// CoarseView_17995912970385
// MI455X (gfx1250) — matrix operand order repaired and checked by reading, not compiled, not run on hardware
//
#include <hip/hip_runtime.h>
#include <hip/hip_bf16.h>
#include <cmath>

#define N_DRUG    708
#define N_TARGET  1512
#define N_DISEASE 5603
#define N_NODES   7823
#define KPAD      7840   // N_NODES padded up to multiple of 32
#define FEAT      1024
#define HID       512
#define OUTD      256

#define KCHUNK    224            // adj GEMM K-chunk staged by TDM (7 wmma k-steps)
#define NCHUNK    (KPAD / KCHUNK) // 35

typedef __bf16 bf16;
typedef __attribute__((ext_vector_type(16))) __bf16 v16bf;
typedef __attribute__((ext_vector_type(8)))  __bf16 v8bf;
typedef __attribute__((ext_vector_type(8)))  float  v8f;
typedef __attribute__((ext_vector_type(4)))  unsigned int u32x4;
typedef __attribute__((ext_vector_type(8)))  int i32x8;
typedef __attribute__((ext_vector_type(4)))  int i32x4;

static __device__ __forceinline__ v16bf cat16(v8bf lo, v8bf hi) {
  return __builtin_shufflevector(lo, hi, 0,1,2,3,4,5,6,7,8,9,10,11,12,13,14,15);
}

// Issue one TDM DMA: 32 rows x KCHUNK cols bf16 tile, global (row stride KPAD
// elements) -> LDS (packed). D# per CDNA5 ISA §8 (groups 0/1; 2D so groups 2/3
// are zero). Caller must be a single wave; completion via TENSORcnt.
static __device__ __forceinline__ void tdm_load_tile(const bf16* gsrc, unsigned ldsOff) {
  unsigned long long ga = (unsigned long long)(size_t)gsrc;
  u32x4 g0;
  g0[0] = 1u;                                   // count=1, no gather, user desc
  g0[1] = ldsOff;                               // lds_addr (bytes)
  g0[2] = (unsigned)ga;                         // global_addr[31:0]
  g0[3] = ((unsigned)(ga >> 32) & 0x01FFFFFFu)  // global_addr[56:32]
          | 0x80000000u;                        // type=2 ("image")
  i32x8 g1;
  g1[0] = 0x00010000;                           // wg_mask=0, data_size=1 (2B)
  g1[1] = (int)((KCHUNK & 0xFFFF) << 16);       // tensor_dim0[15:0] (== tile: no OOB)
  g1[2] = (int)((KCHUNK >> 16) | (32u << 16));  // tensor_dim0[31:16] | tensor_dim1[15:0]
  g1[3] = (int)(0u | ((unsigned)KCHUNK << 16)); // tensor_dim1[31:16] | tile_dim0
  g1[4] = 32;                                   // tile_dim1=32 rows, tile_dim2=0
  g1[5] = KPAD;                                 // tensor_dim0_stride[31:0] (elements)
  g1[6] = 0;                                    // stride[47:32] | dim1_stride[15:0]
  g1[7] = 0;                                    // dim1_stride[47:16]
  i32x4 z4;
  z4[0] = 0; z4[1] = 0; z4[2] = 0; z4[3] = 0;
  i32x8 z8;
  z8[0] = 0; z8[1] = 0; z8[2] = 0; z8[3] = 0;
  z8[4] = 0; z8[5] = 0; z8[6] = 0; z8[7] = 0;
  __builtin_amdgcn_tensor_load_to_lds(g0, g1, z4, z4, z8, 0);
}

// ---------------------------------------------------------------- utilities
__global__ void zero_f32_kernel(float* __restrict__ p, long n) {
  long i = (long)blockIdx.x * blockDim.x + threadIdx.x;
  if (i < n) p[i] = 0.0f;
}

__global__ void cvt_f32_bf16_kernel(const float* __restrict__ in,
                                    bf16* __restrict__ out, long n) {
  long i = (long)blockIdx.x * blockDim.x + threadIdx.x;
  if (i < n) out[i] = (bf16)in[i];
}

// in[K][N] fp32 -> out[N][K] bf16  (B^T form, K contiguous)
__global__ void cvt_transpose_kernel(const float* __restrict__ in,
                                     bf16* __restrict__ out, int K, int N) {
  long j = (long)blockIdx.x * blockDim.x + threadIdx.x;
  if (j >= (long)K * N) return;
  int n = (int)(j / K);
  int k = (int)(j % K);
  out[j] = (bf16)in[(size_t)k * N + n];
}

// adj = 0.5*(adjs[0]+adjs[1]) -> bf16 [KPAD][KPAD]; pad rows/cols zero-filled
__global__ void adj_mean_kernel(const float* __restrict__ adjs,
                                bf16* __restrict__ out) {
  int col = blockIdx.x * blockDim.x + threadIdx.x;
  int row = blockIdx.y;
  if (col >= KPAD) return;
  float v = 0.0f;
  if (col < N_NODES && row < N_NODES) {
    size_t j = (size_t)row * N_NODES + col;
    v = 0.5f * (adjs[j] + adjs[j + (size_t)N_NODES * N_NODES]);
  }
  out[(size_t)row * KPAD + col] = (bf16)v;
}

// ---------------------------------------------------------------- adapter
// out = elu( relu(LN(x@W^T + b)*g + beta) + x@Wr^T + br )
// block = 256 thr = 8 waves; 16 rows x 512 cols; wave w -> cols [64w,64w+64)
__global__ __launch_bounds__(256) void adapter_kernel(
    const bf16* __restrict__ Xb, int Mseg,
    const bf16* __restrict__ Wb, const bf16* __restrict__ Wrb,
    const float* __restrict__ b, const float* __restrict__ g,
    const float* __restrict__ beta, const float* __restrict__ br,
    bf16* __restrict__ xall, int outBase) {
  __shared__ float sy[16][HID];
  __shared__ float red[16][16][2];
  __shared__ float mv[16][2];

  const int wave  = threadIdx.x >> 5;
  const int lane  = threadIdx.x & 31;
  const int lhalf = lane & 15;
  const bool hi   = lane >= 16;
  const int m0    = blockIdx.x * 16;
  const int n0    = wave * 64;

  int rowA = m0 + lhalf;
  if (rowA >= Mseg) rowA = Mseg - 1;
  const bf16* aRow = Xb + (size_t)rowA * FEAT + (hi ? 16 : 0);

  const bf16* wPtr[4];
  const bf16* rPtr[4];
#pragma unroll
  for (int t = 0; t < 4; ++t) {
    int col = n0 + t * 16 + lhalf;
    wPtr[t] = Wb  + (size_t)col * FEAT + (hi ? 16 : 0);
    rPtr[t] = Wrb + (size_t)col * FEAT + (hi ? 16 : 0);
  }

  v8f accY[4], accR[4];
#pragma unroll
  for (int t = 0; t < 4; ++t)
#pragma unroll
    for (int r = 0; r < 8; ++r) { accY[t][r] = 0.0f; accR[t][r] = 0.0f; }

  for (int k = 0; k < FEAT; k += 32) {
    v16bf av = cat16(*(const v8bf*)(aRow + k), *(const v8bf*)(aRow + k + 8));
#pragma unroll
    for (int t = 0; t < 4; ++t) {
      v16bf wv = cat16(*(const v8bf*)(wPtr[t] + k), *(const v8bf*)(wPtr[t] + k + 8));
      accY[t] = __builtin_amdgcn_wmma_f32_16x16x32_bf16(false, av, false, wv,
                                                        (short)0, accY[t], false, false);
      v16bf rv = cat16(*(const v8bf*)(rPtr[t] + k), *(const v8bf*)(rPtr[t] + k + 8));
      accR[t] = __builtin_amdgcn_wmma_f32_16x16x32_bf16(false, av, false, rv,
                                                        (short)0, accR[t], false, false);
    }
  }

  const int mB = hi ? 8 : 0;
#pragma unroll
  for (int t = 0; t < 4; ++t) {
    int n = n0 + t * 16 + lhalf;
#pragma unroll
    for (int r = 0; r < 8; ++r) sy[mB + r][n] = accY[t][r] + b[n];
  }
  __syncthreads();

  {
    int row = threadIdx.x >> 4, part = threadIdx.x & 15;
    float s = 0.0f, s2 = 0.0f;
    for (int c = part; c < HID; c += 16) { float v = sy[row][c]; s += v; s2 += v * v; }
    red[row][part][0] = s; red[row][part][1] = s2;
  }
  __syncthreads();
  if (threadIdx.x < 16) {
    float s = 0.0f, s2 = 0.0f;
    for (int i = 0; i < 16; ++i) { s += red[threadIdx.x][i][0]; s2 += red[threadIdx.x][i][1]; }
    float mean = s * (1.0f / HID);
    float var  = s2 * (1.0f / HID) - mean * mean;
    mv[threadIdx.x][0] = mean;
    mv[threadIdx.x][1] = rsqrtf(var + 1e-5f);
  }
  __syncthreads();

#pragma unroll
  for (int t = 0; t < 4; ++t) {
    int n = n0 + t * 16 + lhalf;
#pragma unroll
    for (int r = 0; r < 8; ++r) {
      int m = mB + r;
      float ln = (sy[m][n] - mv[m][0]) * mv[m][1] * g[n] + beta[n];
      float v = (ln > 0.0f ? ln : 0.0f) + accR[t][r] + br[n];
      float e = v > 0.0f ? v : expm1f(v);
      int grow = m0 + m;
      if (grow < Mseg)
        xall[(size_t)(outBase + grow) * HID + n] = (bf16)e;
    }
  }
}

// ---------------------------------------------------------------- small GEMM
// C[M,256] = A[M,K] @ B (BT[256][K]); write bf16 transposed outB[n*ldo+m]
__global__ __launch_bounds__(128) void gemm_small_wmma(
    const bf16* __restrict__ A, int lda,
    const bf16* __restrict__ BT, int ldb,
    int M, int K,
    bf16* __restrict__ outB, int ldo) {
  const int wave  = threadIdx.x >> 5;
  const int lane  = threadIdx.x & 31;
  const int lhalf = lane & 15;
  const bool hi   = lane >= 16;
  const int m0    = blockIdx.x * 16;
  const int n0    = wave * 64;

  int rowA = m0 + lhalf;
  if (rowA >= M) rowA = M - 1;
  const bf16* aRow = A + (size_t)rowA * lda + (hi ? 16 : 0);

  const bf16* bRow[4];
#pragma unroll
  for (int t = 0; t < 4; ++t)
    bRow[t] = BT + (size_t)(n0 + t * 16 + lhalf) * ldb + (hi ? 16 : 0);

  v8f acc[4];
#pragma unroll
  for (int t = 0; t < 4; ++t)
#pragma unroll
    for (int r = 0; r < 8; ++r) acc[t][r] = 0.0f;

  for (int k = 0; k < K; k += 32) {
    v16bf av = cat16(*(const v8bf*)(aRow + k), *(const v8bf*)(aRow + k + 8));
#pragma unroll
    for (int t = 0; t < 4; ++t) {
      v16bf bv = cat16(*(const v8bf*)(bRow[t] + k), *(const v8bf*)(bRow[t] + k + 8));
      acc[t] = __builtin_amdgcn_wmma_f32_16x16x32_bf16(false, av, false, bv,
                                                       (short)0, acc[t], false, false);
    }
  }

  const int mBase = m0 + (hi ? 8 : 0);
#pragma unroll
  for (int t = 0; t < 4; ++t) {
    int n = n0 + t * 16 + lhalf;
#pragma unroll
    for (int r = 0; r < 8; ++r) {
      int m = mBase + r;
      if (m < M) outB[(size_t)n * ldo + m] = (bf16)acc[t][r];
    }
  }
}

// ---------------------------------------------------------------- adj GEMM
// C[KPAD,256] = adj[KPAD][KPAD] @ B (BT[256][KPAD]), A staged via TDM into
// double-buffered LDS. block = 256 thr = 8 waves; M-tile 32, N=256.
// MODE 1: PReLU(a) -> bf16 row-major outB[m*256+n]
// MODE 2: fp32 Z[m*256+n] + atomic sum/sumsq into stats[0..1]
template <int MODE>
__global__ __launch_bounds__(256) void gemm_adj_tdm(
    const bf16* __restrict__ A,      // [KPAD][KPAD]
    const bf16* __restrict__ BT,     // [256][KPAD]
    bf16* __restrict__ outB,
    float* __restrict__ Zout,
    const float* __restrict__ prelu_a,
    float* __restrict__ stats) {
  __shared__ bf16 tile[2][32][KCHUNK];

  const int tid   = threadIdx.x;
  const int wave  = tid >> 5;
  const int lane  = tid & 31;
  const int lhalf = lane & 15;
  const bool hi   = lane >= 16;
  const int m0    = blockIdx.x * 32;
  const int waveM = (wave >> 2) * 16;   // 0 or 16
  const int n0    = (wave & 3) * 64;

  const bf16* bRow[4];
#pragma unroll
  for (int t = 0; t < 4; ++t)
    bRow[t] = BT + (size_t)(n0 + t * 16 + lhalf) * KPAD + (hi ? 16 : 0);

  v8f acc[4];
#pragma unroll
  for (int t = 0; t < 4; ++t)
#pragma unroll
    for (int r = 0; r < 8; ++r) acc[t][r] = 0.0f;

  const bf16* gBase = A + (size_t)m0 * KPAD;
  const unsigned ldsOff0 = (unsigned)(size_t)(&tile[0][0][0]);
  const unsigned ldsOff1 = (unsigned)(size_t)(&tile[1][0][0]);

  if (tid < 32) tdm_load_tile(gBase, ldsOff0);           // chunk 0 -> buf 0

  for (int c = 0; c < NCHUNK; ++c) {
    if (tid < 32) {
      if (c + 1 < NCHUNK) {
        tdm_load_tile(gBase + (size_t)(c + 1) * KCHUNK,
                      ((c + 1) & 1) ? ldsOff1 : ldsOff0);
        __builtin_amdgcn_s_wait_tensorcnt(1);            // chunk c done
      } else {
        __builtin_amdgcn_s_wait_tensorcnt(0);
      }
    }
    __syncthreads();                                     // tile[c&1] ready

    const bf16* at = &tile[c & 1][waveM + lhalf][hi ? 16 : 0];
    const int kg = c * KCHUNK;
#pragma unroll
    for (int k = 0; k < KCHUNK; k += 32) {
      v16bf av = cat16(*(const v8bf*)(at + k), *(const v8bf*)(at + k + 8));
#pragma unroll
      for (int t = 0; t < 4; ++t) {
        v16bf bv = cat16(*(const v8bf*)(bRow[t] + kg + k),
                         *(const v8bf*)(bRow[t] + kg + k + 8));
        acc[t] = __builtin_amdgcn_wmma_f32_16x16x32_bf16(false, av, false, bv,
                                                         (short)0, acc[t], false, false);
      }
    }
    __syncthreads();                                     // tile[c&1] reusable
  }

  const int mBase = m0 + waveM + (hi ? 8 : 0);
  if constexpr (MODE == 1) {
    float a = prelu_a[0];
#pragma unroll
    for (int t = 0; t < 4; ++t) {
      int n = n0 + t * 16 + lhalf;
#pragma unroll
      for (int r = 0; r < 8; ++r) {
        int m = mBase + r;
        float v = acc[t][r];
        v = v >= 0.0f ? v : a * v;
        if (m < N_NODES) outB[(size_t)m * OUTD + n] = (bf16)v;
      }
    }
  } else {
    float s = 0.0f, s2 = 0.0f;
#pragma unroll
    for (int t = 0; t < 4; ++t) {
      int n = n0 + t * 16 + lhalf;
#pragma unroll
      for (int r = 0; r < 8; ++r) {
        int m = mBase + r;
        float v = acc[t][r];
        if (m < N_NODES) {
          Zout[(size_t)m * OUTD + n] = v;
          s += v; s2 += v * v;
        }
      }
    }
#pragma unroll
    for (int off = 16; off >= 1; off >>= 1) {
      s  += __shfl_xor(s, off, 32);
      s2 += __shfl_xor(s2, off, 32);
    }
    if (lane == 0) { atomicAdd(&stats[0], s); atomicAdd(&stats[1], s2); }
  }
}

// ---------------------------------------------------------------- finalize
__global__ __launch_bounds__(256) void finalize_kernel(
    const float* __restrict__ Z, const float* __restrict__ stats,
    int rowStart, float* __restrict__ out) {
  const int row = rowStart + blockIdx.x;
  const int c = threadIdx.x;
  const float inv = 1.0f / ((float)N_NODES * (float)OUTD);
  float mean = stats[0] * inv;
  float var  = stats[1] * inv - mean * mean;
  float rstd = rsqrtf(var + 1e-5f);
  float v = (Z[(size_t)row * OUTD + c] - mean) * rstd;

  float s = v * v;
#pragma unroll
  for (int off = 16; off >= 1; off >>= 1) s += __shfl_xor(s, off, 32);
  __shared__ float wsum[8];
  if ((threadIdx.x & 31) == 0) wsum[threadIdx.x >> 5] = s;
  __syncthreads();
  float tot = 0.0f;
#pragma unroll
  for (int w = 0; w < 8; ++w) tot += wsum[w];
  out[(size_t)blockIdx.x * OUTD + c] = v / (sqrtf(tot) + 1e-8f);
}

// ---------------------------------------------------------------- host
static inline void* ws_alloc(char*& p, size_t bytes) {
  void* r = (void*)p;
  p += (bytes + 255) & ~(size_t)255;
  return r;
}

extern "C" void kernel_launch(void* const* d_in, const int* in_sizes, int n_in,
                              void* d_out, int out_size, void* d_ws, size_t ws_size,
                              hipStream_t stream) {
  const float* feats[3]   = {(const float*)d_in[0], (const float*)d_in[1], (const float*)d_in[2]};
  const int    segRows[3] = {N_DRUG, N_TARGET, N_DISEASE};
  const int    segBase[3] = {0, N_DRUG, N_DRUG + N_TARGET};
  const float* adjs[2]    = {(const float*)d_in[3], (const float*)d_in[4]};
  const float* gW1[2]     = {(const float*)d_in[23], (const float*)d_in[25]};
  const float* gW2[2]     = {(const float*)d_in[24], (const float*)d_in[26]};
  const float* gA[2]      = {(const float*)d_in[27], (const float*)d_in[28]};
  float* out = (float*)d_out;

  char* wp = (char*)d_ws;
  bf16*  featb = (bf16*)ws_alloc(wp, (size_t)N_NODES * FEAT * 2);
  bf16*  xall  = (bf16*)ws_alloc(wp, (size_t)N_NODES * HID * 2);
  bf16*  adjbf = (bf16*)ws_alloc(wp, (size_t)KPAD * KPAD * 2);   // padded rows too
  bf16*  yT    = (bf16*)ws_alloc(wp, (size_t)OUTD * KPAD * 2);
  bf16*  hbuf  = (bf16*)ws_alloc(wp, (size_t)N_NODES * OUTD * 2);
  float* Zbuf  = (float*)ws_alloc(wp, (size_t)N_NODES * OUTD * 4);
  bf16*  aWb[3]; bf16* aWrb[3];
  for (int i = 0; i < 3; ++i) {
    aWb[i]  = (bf16*)ws_alloc(wp, (size_t)HID * FEAT * 2);
    aWrb[i] = (bf16*)ws_alloc(wp, (size_t)HID * FEAT * 2);
  }
  bf16* W1T[2]; bf16* W2T[2];
  for (int g = 0; g < 2; ++g) {
    W1T[g] = (bf16*)ws_alloc(wp, (size_t)OUTD * HID * 2);
    W2T[g] = (bf16*)ws_alloc(wp, (size_t)OUTD * OUTD * 2);
  }
  float* stats = (float*)ws_alloc(wp, 256);

  // --- conversions ---
  for (int s = 0; s < 3; ++s) {
    long n = (long)segRows[s] * FEAT;
    cvt_f32_bf16_kernel<<<(int)((n + 255) / 256), 256, 0, stream>>>(
        feats[s], featb + (size_t)segBase[s] * FEAT, n);
  }
  for (int s = 0; s < 3; ++s) {
    const int base = 5 + s * 6;
    long n = (long)HID * FEAT;
    cvt_f32_bf16_kernel<<<(int)((n + 255) / 256), 256, 0, stream>>>(
        (const float*)d_in[base + 0], aWb[s], n);          // W
    cvt_f32_bf16_kernel<<<(int)((n + 255) / 256), 256, 0, stream>>>(
        (const float*)d_in[base + 4], aWrb[s], n);         // Wr
  }
  for (int g = 0; g < 2; ++g) {
    long n1 = (long)HID * OUTD, n2 = (long)OUTD * OUTD;
    cvt_transpose_kernel<<<(int)((n1 + 255) / 256), 256, 0, stream>>>(gW1[g], W1T[g], HID, OUTD);
    cvt_transpose_kernel<<<(int)((n2 + 255) / 256), 256, 0, stream>>>(gW2[g], W2T[g], OUTD, OUTD);
  }

  // --- adapters -> x_all ---
  for (int s = 0; s < 3; ++s) {
    const int base = 5 + s * 6;
    adapter_kernel<<<(segRows[s] + 15) / 16, 256, 0, stream>>>(
        featb + (size_t)segBase[s] * FEAT, segRows[s],
        aWb[s], aWrb[s],
        (const float*)d_in[base + 1], (const float*)d_in[base + 2],
        (const float*)d_in[base + 3], (const float*)d_in[base + 5],
        xall, segBase[s]);
  }

  const int smallBlocks = (N_NODES + 15) / 16;
  const int adjBlocks   = KPAD / 32;                 // 245
  const long yTn = (long)OUTD * KPAD / 2;            // fp32 words in yT buffer

  for (int g = 0; g < 2; ++g) {
    // adj = mean of two adjs, bf16, fully padded [KPAD][KPAD]
    dim3 agrid((KPAD + 255) / 256, KPAD);
    adj_mean_kernel<<<agrid, 256, 0, stream>>>(adjs[g], adjbf);

    // y1T = (x_all @ W1)^T   [256][KPAD]
    zero_f32_kernel<<<(int)((yTn + 255) / 256), 256, 0, stream>>>((float*)yT, yTn);
    gemm_small_wmma<<<smallBlocks, 128, 0, stream>>>(xall, HID, W1T[g], HID,
                                                     N_NODES, HID, yT, KPAD);
    // h = prelu(adj @ y1)    [N_NODES][256] row-major
    gemm_adj_tdm<1><<<adjBlocks, 256, 0, stream>>>(adjbf, yT, hbuf,
                                                   nullptr, gA[g], nullptr);
    // y2T = (h @ W2)^T
    zero_f32_kernel<<<(int)((yTn + 255) / 256), 256, 0, stream>>>((float*)yT, yTn);
    gemm_small_wmma<<<smallBlocks, 128, 0, stream>>>(hbuf, OUTD, W2T[g], OUTD,
                                                     N_NODES, OUTD, yT, KPAD);
    // Z = adj @ y2 (fp32) + global stats
    zero_f32_kernel<<<1, 256, 0, stream>>>(stats, 2);
    gemm_adj_tdm<2><<<adjBlocks, 256, 0, stream>>>(adjbf, yT, nullptr,
                                                   Zbuf, nullptr, stats);
    // slice + whole-tensor LN + row L2 normalize
    const int rowStart = (g == 0) ? 0 : N_DRUG;
    const int nRows    = (g == 0) ? N_DRUG : N_TARGET;
    float* outSeg      = out + (size_t)((g == 0) ? 0 : N_DRUG * OUTD);
    finalize_kernel<<<nRows, 256, 0, stream>>>(Zbuf, stats, rowStart, outSeg);
  }
  (void)in_sizes; (void)n_in; (void)out_size; (void)ws_size;
}
